// TimeLLM_39968965657193
// MI455X (gfx1250) — matrix operand order repaired and checked by reading, not compiled, not run on hardware
//
#include <hip/hip_runtime.h>

typedef __bf16 bhalf;
typedef __attribute__((ext_vector_type(16))) __bf16 v16bf;
typedef __attribute__((ext_vector_type(8)))  float  v8f;

#define B_   4
#define L_   128
#define DM   256
#define DW   768
#define VV   50257
#define H_   8
#define E_   32
#define BL   (B_ * L_)          // 512 rows of Q / output
#define VC   64                 // vocab rows per chunk
#define W_STR   776             // ldsW stride (bf16) : 768 + 8 pad (bank spread)
#define KV_STR  264             // ldsK stride (bf16) : 256 + 8 pad
#define VT_STR  72              // ldsVT stride (bf16): 64 + 8 pad (V chunk, transposed [e][v])
#define S_STR   132             // ldsS stride (f32)  : 128 + 4 pad
#define P_STR   68              // ldsP stride (bf16) : 64 + 4 pad

// LDS region offsets (bytes)
#define OFF_K   ((size_t)VC * W_STR * 2)                 // 99328
#define OFF_VT  (OFF_K + (size_t)VC * KV_STR * 2)        // 133120
#define OFF_P   ((size_t)VC * S_STR * 4)                 // 33792 (inside word region)
#define LDS_BYTES (OFF_VT + (size_t)DM * VT_STR * 2)     // 169984

// ---------------- WMMA operand loaders (ISA 7.12.2 layouts, wave32) ----------

// A-matrix 16x32 bf16, source row-major [m][k] with given stride.
// lane<16: M=lane, K = 0..7 then 16..23 ; lane>=16: M=lane-16, K = 8..15 then 24..31
__device__ __forceinline__ v16bf load_a16x32(const bhalf* base, int stride, int lane, int k0)
{
    const int m  = lane & 15;
    const int kb = (lane >> 4) * 8;
    const bhalf* row = base + (size_t)m * stride + k0 + kb;
    v16bf a;
#pragma unroll
    for (int j = 0; j < 4; ++j) { a[2*j] = row[2*j]; a[2*j+1] = row[2*j+1]; }
#pragma unroll
    for (int j = 4; j < 8; ++j) { int k = 16 + 2*(j-4); a[2*j] = row[k]; a[2*j+1] = row[k+1]; }
    return a;
}

// B-matrix 32x16 bf16 from TRANSPOSED storage [n][k] (pairs contiguous along k).
// lane<16: N=lane, K = 0..7 then 16..23 ; lane>=16: N=lane-16, K = 8..15 then 24..31
__device__ __forceinline__ v16bf load_b32x16_t(const bhalf* base, int stride, int lane, int k0)
{
    const int n  = lane & 15;
    const int kb = (lane >> 4) * 8;
    const bhalf* p = base + (size_t)n * stride + k0 + kb;
    v16bf b;
#pragma unroll
    for (int j = 0; j < 8; ++j) { b[j] = p[j]; b[8+j] = p[16+j]; }
    return b;
}

__device__ __forceinline__ v8f wmma_bf16(v16bf a, v16bf b, v8f c)
{
    return __builtin_amdgcn_wmma_f32_16x16x32_bf16(false, a, false, b, (short)0, c, false, false);
}

// ---------------- prep kernels ------------------------------------------------

// Wk/Wv (f32 [dw][dm]) -> bf16 transposed [dm][dw]
__global__ void prep_weights(const float* __restrict__ Wk, const float* __restrict__ Wv,
                             bhalf* __restrict__ WkT, bhalf* __restrict__ WvT)
{
    int i = blockIdx.x * 256 + threadIdx.x;          // 2*768*256 threads
    int m = (i >= DW * DM) ? 1 : 0;
    int j = i - m * DW * DM;
    int r = j >> 8;           // dw row
    int c = j & 255;          // dm col
    const float* src = m ? Wv : Wk;
    bhalf* dst       = m ? WvT : WkT;
    dst[(size_t)c * DW + r] = (bhalf)src[(size_t)r * DM + c];
}

// Q = TS @ Wq + bq  (f32 -> bf16), and zero the reprog accumulator.
__global__ void q_proj_zero(const float* __restrict__ ts, const float* __restrict__ Wq,
                            const float* __restrict__ bq, bhalf* __restrict__ Qbf,
                            float* __restrict__ Racc)
{
    int i = blockIdx.x * 256 + threadIdx.x;          // 512*256 threads
    int row = i >> 8, col = i & 255;
    float s = bq[col];
    for (int k = 0; k < DM; ++k)
        s += ts[(size_t)row * DM + k] * Wq[(size_t)k * DM + col];
    Qbf[i]  = (bhalf)s;
    Racc[i] = 0.0f;
}

// ---------------- main fused kernel ------------------------------------------

__launch_bounds__(256)
__global__ void fused_reprog(const float* __restrict__ wordEmb,
                             const bhalf* __restrict__ WkT, const bhalf* __restrict__ WvT,
                             const float* __restrict__ bk,  const float* __restrict__ bv,
                             const bhalf* __restrict__ Qbf, float* __restrict__ Racc)
{
    extern __shared__ char smem[];
    bhalf* ldsW  = (bhalf*)smem;                 // [VC][W_STR]   word chunk (bf16)
    bhalf* ldsK  = (bhalf*)(smem + OFF_K);       // [VC][KV_STR]  K chunk, [v][e]
    bhalf* ldsVT = (bhalf*)(smem + OFF_VT);      // [DM][VT_STR]  V chunk, transposed [e][v]
    // S and P alias the (dead after projection) word-chunk region
    float* ldsS  = (float*)smem;                 // [VC][S_STR]   scores f32
    bhalf* ldsP  = (bhalf*)(smem + OFF_P);       // [L_][P_STR]   probs bf16, [l][v]

    const int tid  = threadIdx.x;
    const int wave = tid >> 5;
    const int lane = tid & 31;
    const int v0   = blockIdx.x * VC;

    // ---- stage vocab chunk (f32 -> bf16) into LDS via float4, zero-pad past V ----
    for (int i = tid; i < VC * (DW / 4); i += 256) {
        int r = i / (DW / 4);
        int c = (i - r * (DW / 4)) * 4;
        int gv = v0 + r;
        float4 x = make_float4(0.f, 0.f, 0.f, 0.f);
        if (gv < VV) x = *(const float4*)(wordEmb + (size_t)gv * DW + c);
        bhalf* w = ldsW + r * W_STR + c;
        w[0] = (bhalf)x.x; w[1] = (bhalf)x.y; w[2] = (bhalf)x.z; w[3] = (bhalf)x.w;
    }
    // prefetch next chunk (one byte per 256B line) -> global_prefetch_b8
    {
        int nv0 = v0 + VC;
        for (int c = tid; c < VC * 12; c += 256) {
            int r = c / 12, off = (c - r * 12) * 64;
            int gv = nv0 + r;
            if (gv < VV) __builtin_prefetch(&wordEmb[(size_t)gv * DW + off], 0, 1);
        }
    }
    __syncthreads();

    // ---- K/V projection: [VC,768] x [768,256] via WMMA, +bias ----
    // 128 tile jobs (2 matrices x 4 v-tiles x 16 n-tiles), 16 per wave.
    // K chunk stored natural [v][e]; V chunk stored transposed [e][v].
    for (int t = 0; t < 16; ++t) {
        int job = wave * 16 + t;
        const bhalf* WT = (job & 64) ? WvT : WkT;
        const float* bs = (job & 64) ? bv  : bk;
        int rem = job & 63;
        int vt = rem >> 4, nt = rem & 15;
        v8f acc = {};
#pragma unroll 4
        for (int k0 = 0; k0 < DW; k0 += 32) {
            v16bf a = load_a16x32(ldsW + vt * 16 * W_STR, W_STR, lane, k0);
            v16bf b = load_b32x16_t(WT + (size_t)(nt * 16) * DW, DW, lane, k0);
            acc = wmma_bf16(a, b, acc);
        }
        int n = lane & 15, mb = (lane >> 4) * 8;
        float bias = bs[nt * 16 + n];
        if (job & 64) {
            // V: transposed store -> 8 consecutive bf16 (merges to ds_store_b128)
            bhalf* dst = ldsVT + (size_t)(nt * 16 + n) * VT_STR + vt * 16 + mb;
#pragma unroll
            for (int r = 0; r < 8; ++r) dst[r] = (bhalf)(acc[r] + bias);
        } else {
            bhalf* dst = ldsK + (size_t)vt * 16 * KV_STR + nt * 16 + n;
#pragma unroll
            for (int r = 0; r < 8; ++r) dst[(mb + r) * KV_STR] = (bhalf)(acc[r] + bias);
        }
    }
    __syncthreads();

    const float scale = 0.17677669529663687f;   // 1/sqrt(E)

    // ---- per (b,h): scores -> softmax over L -> reprog partial ----
    for (int bh = 0; bh < B_ * H_; ++bh) {
        int bb = bh >> 3, h = bh & 7;

        // scores S[v, l] = K_chunk[v,:e] . Q[b,l,h,:e] ; 32 tiles, 4/wave
#pragma unroll
        for (int t = 0; t < 4; ++t) {
            int tile = wave * 4 + t;
            int vt = tile >> 3, lt = tile & 7;
            v16bf a = load_a16x32(ldsK + vt * 16 * KV_STR + h * E_, KV_STR, lane, 0);
            v16bf b = load_b32x16_t(Qbf + (size_t)(bb * L_ + lt * 16) * DM + h * E_, DM, lane, 0);
            v8f c = {};
            c = wmma_bf16(a, b, c);
            int n = lane & 15, mb = (lane >> 4) * 8;
#pragma unroll
            for (int r = 0; r < 8; ++r)
                ldsS[(vt * 16 + mb + r) * S_STR + lt * 16 + n] = c[r];
        }
        __syncthreads();

        // softmax over l: 4 threads per row (32 elems each), combine via shfl_xor
        {
            int row = tid >> 2;          // 0..63 (vocab row in chunk)
            int seg = tid & 3;           // 0..3  (32-wide segment of l)
            const float* srow = ldsS + row * S_STR + seg * 32;
            float mx = -3.0e38f;
#pragma unroll
            for (int l = 0; l < 32; ++l) mx = fmaxf(mx, srow[l]);
            mx = fmaxf(mx, __shfl_xor(mx, 1, 32));
            mx = fmaxf(mx, __shfl_xor(mx, 2, 32));
            float sum = 0.0f;
#pragma unroll
            for (int l = 0; l < 32; ++l) sum += __expf(scale * (srow[l] - mx));
            sum += __shfl_xor(sum, 1, 32);
            sum += __shfl_xor(sum, 2, 32);
            float inv = (v0 + row < VV) ? (1.0f / sum) : 0.0f;   // mask padded rows
#pragma unroll
            for (int l = 0; l < 32; ++l)
                ldsP[(seg * 32 + l) * P_STR + row] = (bhalf)(__expf(scale * (srow[l] - mx)) * inv);
        }
        __syncthreads();

        // reprog partial R[l, e] = sum_v P[v,l] * V_chunk[v,e] ; 16 tiles, 2/wave
        // A = P [l][v] (natural), B = V^T [e][v] (transposed storage -> contiguous pairs)
#pragma unroll
        for (int t = 0; t < 2; ++t) {
            int tile = wave * 2 + t;
            int lt = tile >> 1, et = tile & 1;
            const bhalf* Bbase = ldsVT + (size_t)(h * E_ + et * 16) * VT_STR;
            v8f c = {};
            c = wmma_bf16(load_a16x32(ldsP + (size_t)lt * 16 * P_STR, P_STR, lane, 0),
                          load_b32x16_t(Bbase, VT_STR, lane, 0), c);
            c = wmma_bf16(load_a16x32(ldsP + (size_t)lt * 16 * P_STR, P_STR, lane, 32),
                          load_b32x16_t(Bbase, VT_STR, lane, 32), c);
            int n = lane & 15, mb = (lane >> 4) * 8;
#pragma unroll
            for (int r = 0; r < 8; ++r) {
                int l = lt * 16 + mb + r;
                atomicAdd(&Racc[((size_t)(bb * L_ + l)) * DM + h * E_ + et * 16 + n], c[r]);
            }
        }
        __syncthreads();
    }
}

// ---------------- epilogue ----------------------------------------------------

__global__ void out_proj(const float* __restrict__ Racc, const float* __restrict__ Wo,
                         const float* __restrict__ bo, float* __restrict__ out)
{
    int i = blockIdx.x * 256 + threadIdx.x;          // 512*256 threads
    int row = i >> 8, col = i & 255;
    float s = bo[col];
    for (int k = 0; k < DM; ++k)
        s += Racc[(size_t)row * DM + k] * Wo[(size_t)k * DM + col];
    out[i] = s;
}

// ---------------- host glue ---------------------------------------------------

extern "C" void kernel_launch(void* const* d_in, const int* in_sizes, int n_in,
                              void* d_out, int out_size, void* d_ws, size_t ws_size,
                              hipStream_t stream)
{
    const float* ts = (const float*)d_in[0];
    const float* we = (const float*)d_in[1];
    const float* Wq = (const float*)d_in[2];
    const float* bq = (const float*)d_in[3];
    const float* Wk = (const float*)d_in[4];
    const float* bk = (const float*)d_in[5];
    const float* Wv = (const float*)d_in[6];
    const float* bv = (const float*)d_in[7];
    const float* Wo = (const float*)d_in[8];
    const float* bo = (const float*)d_in[9];
    float* out = (float*)d_out;

    char* ws = (char*)d_ws;
    bhalf* Qbf  = (bhalf*)ws;                                        // 512*256 bf16
    bhalf* WkT  = (bhalf*)(ws + (size_t)BL * DM * 2);                // 256*768 bf16
    bhalf* WvT  = (bhalf*)(ws + (size_t)BL * DM * 2 + (size_t)DW * DM * 2);
    float* Racc = (float*)(ws + (size_t)BL * DM * 2 + 2 * (size_t)DW * DM * 2); // 512*256 f32

    prep_weights<<<(2 * DW * DM) / 256, 256, 0, stream>>>(Wk, Wv, WkT, WvT);
    q_proj_zero<<<(BL * DM) / 256, 256, 0, stream>>>(ts, Wq, bq, Qbf, Racc);

    const int nchunks = (VV + VC - 1) / VC;
    fused_reprog<<<nchunks, 256, LDS_BYTES, stream>>>(we, WkT, WvT, bk, bv, Qbf, Racc);

    out_proj<<<(BL * DM) / 256, 256, 0, stream>>>(Racc, Wo, bo, out);
}
